// ContextQueryAttentionLayer_8598524527150
// MI455X (gfx1250) — matrix operand order repaired and checked by reading, not compiled, not run on hardware
//
#include <hip/hip_runtime.h>
#include <hip/hip_bf16.h>

// ---------------------------------------------------------------------------
// Context-Query Attention (BiDAF) for MI455X / gfx1250, wave32 + bf16 WMMA.
// B=32, N=1024, M=256, D=512.
// ---------------------------------------------------------------------------

typedef __bf16 bf16_t;
typedef bf16_t v16bf __attribute__((ext_vector_type(16)));
typedef bf16_t v8bf  __attribute__((ext_vector_type(8)));
typedef float  v8f   __attribute__((ext_vector_type(8)));
typedef float  v4f   __attribute__((ext_vector_type(4)));

constexpr int kB = 32;
constexpr int kN = 1024;
constexpr int kM = 256;
constexpr int kD = 512;
constexpr float kNegBig = -1.0e9f;

// ---------------------------------------------------------------------------
// Wave helpers (wave32)
// ---------------------------------------------------------------------------
__device__ __forceinline__ float wave_max(float x) {
#pragma unroll
  for (int o = 16; o > 0; o >>= 1) x = fmaxf(x, __shfl_xor(x, o, 32));
  return x;
}
__device__ __forceinline__ float wave_sum(float x) {
#pragma unroll
  for (int o = 16; o > 0; o >>= 1) x += __shfl_xor(x, o, 32);
  return x;
}

// ---------------------------------------------------------------------------
// WMMA fragment loaders (16x16x32 bf16, wave32; A and B share a contiguous k order, not the ISA table order)
//
// A (16xK, row-major source, K contiguous): lane holds row = r0 + (lane&15);
//   elems 0..15 -> K = kb + half*16 + e
// B (Kx16, source in [col][K] layout, K contiguous): lane holds
//   col = c0 + (lane&15); elems 0..15 -> K = kb + half*16 + e
// C/D (16x16 f32): elem i -> row = r0 + half*8 + i, col = c0 + (lane&15)
// ---------------------------------------------------------------------------
__device__ __forceinline__ v16bf load_a_frag(const bf16_t* __restrict__ src,
                                             int ld, int r0, int kb, int lane) {
  const int r = r0 + (lane & 15);
  const int half = lane >> 4;
  const bf16_t* p = src + (size_t)r * ld + kb + half * 16;
  v8bf lo = *(const v8bf*)p;          // 16B
  v8bf hi = *(const v8bf*)(p + 8);    // 16B
  v16bf out;
#pragma unroll
  for (int i = 0; i < 8; ++i) { out[i] = lo[i]; out[i + 8] = hi[i]; }
  return out;
}

__device__ __forceinline__ v16bf load_b_frag(const bf16_t* __restrict__ srcT,
                                             int ld, int c0, int kb, int lane) {
  const int c = c0 + (lane & 15);
  const int half = lane >> 4;
  return *(const v16bf*)(srcT + (size_t)c * ld + kb + half * 16);  // 32B
}

#define WMMA_BF16(a, b, c) \
  __builtin_amdgcn_wmma_f32_16x16x32_bf16(false, (a), false, (b), (short)0, (c), false, false)

// ---------------------------------------------------------------------------
// K0a: per context row: cw = C . w_c  (w[D:2D]), plain bf16 copy of C.
// One wave per row, 8 rows / 256-thread block.
// ---------------------------------------------------------------------------
__global__ void prep_ctx_kernel(const float* __restrict__ context,
                                const float* __restrict__ w,
                                float* __restrict__ cw,
                                bf16_t* __restrict__ Cb) {
  const int lane = threadIdx.x & 31;
  const int row = blockIdx.x * 8 + (threadIdx.x >> 5);  // [0, B*N)
  const float* cr = context + (size_t)row * kD;
  const float* wc = w + kD;
  bf16_t* cb = Cb + (size_t)row * kD;
  float dot = 0.f;
#pragma unroll
  for (int i = 0; i < kD / 32; ++i) {
    const int d = lane + 32 * i;
    const float v = cr[d];
    dot += v * wc[d];
    cb[d] = (bf16_t)v;
  }
  dot = wave_sum(dot);
  if (lane == 0) cw[row] = dot;
}

// ---------------------------------------------------------------------------
// K0b: per query row: qw = Q . w_q (w[0:D]), bf16 copy of Q*diag(w_m) (w[2D:3D]).
// ---------------------------------------------------------------------------
__global__ void prep_query_kernel(const float* __restrict__ query,
                                  const float* __restrict__ w,
                                  float* __restrict__ qw,
                                  bf16_t* __restrict__ Qmb) {
  const int lane = threadIdx.x & 31;
  const int row = blockIdx.x * 8 + (threadIdx.x >> 5);  // [0, B*M)
  const float* qr = query + (size_t)row * kD;
  const float* wq = w;
  const float* wm = w + 2 * kD;
  bf16_t* qb = Qmb + (size_t)row * kD;
  float dot = 0.f;
#pragma unroll
  for (int i = 0; i < kD / 32; ++i) {
    const int d = lane + 32 * i;
    const float v = qr[d];
    dot += v * wq[d];
    qb[d] = (bf16_t)(v * wm[d]);
  }
  dot = wave_sum(dot);
  if (lane == 0) qw[row] = dot;
}

// ---------------------------------------------------------------------------
// LDS-tiled transpose: [rows, cols] (f32 or bf16 in) -> [cols, rows] bf16 out.
// grid = (cols/32, rows/32, B), block = (32, 8)
// ---------------------------------------------------------------------------
template <typename Tin>
__global__ void transpose_to_bf16(const Tin* __restrict__ in,
                                  bf16_t* __restrict__ out,
                                  int rows, int cols) {
  __shared__ bf16_t tile[32][33];
  const int b = blockIdx.z;
  in += (size_t)b * rows * cols;
  out += (size_t)b * rows * cols;
  const int c0 = blockIdx.x * 32, r0 = blockIdx.y * 32;
  const int x = threadIdx.x, y = threadIdx.y;
#pragma unroll
  for (int k = 0; k < 4; ++k)
    tile[y + 8 * k][x] = (bf16_t)in[(size_t)(r0 + y + 8 * k) * cols + c0 + x];
  __syncthreads();
#pragma unroll
  for (int k = 0; k < 4; ++k)
    out[(size_t)(c0 + y + 8 * k) * rows + r0 + x] = tile[x][y + 8 * k];
}

// ---------------------------------------------------------------------------
// K1: S[b,n,m] = sum_d Cb[n,d]*Qmb[m,d] + cw[n] + qw[m]
// Wave tile 32(n) x 64(m), K = D = 512, 8 WMMAs / K-step.
// 128 threads = 4 waves / block; waves total = B*(N/32)*(M/64) = 4096.
// ---------------------------------------------------------------------------
__global__ __launch_bounds__(128) void s_gemm_kernel(
    const bf16_t* __restrict__ Cb, const bf16_t* __restrict__ Qmb,
    const float* __restrict__ cw, const float* __restrict__ qw,
    float* __restrict__ S) {
  const int lane = threadIdx.x & 31;
  const int wave = blockIdx.x * 4 + (threadIdx.x >> 5);
  const int b = wave >> 7;           // 128 waves per batch
  const int rem = wave & 127;
  const int r0 = (rem >> 2) * 32;    // 32 n-tiles
  const int c0 = (rem & 3) * 64;     // 4 m-tiles
  const bf16_t* A = Cb + (size_t)b * kN * kD;
  const bf16_t* Bq = Qmb + (size_t)b * kM * kD;

  v8f acc[2][4] = {};
  for (int kb = 0; kb < kD; kb += 32) {
    const v16bf a0 = load_a_frag(A, kD, r0, kb, lane);
    const v16bf a1 = load_a_frag(A, kD, r0 + 16, kb, lane);
#pragma unroll
    for (int j = 0; j < 4; ++j) {
      const v16bf bb = load_b_frag(Bq, kD, c0 + 16 * j, kb, lane);
      acc[0][j] = WMMA_BF16(a0, bb, acc[0][j]);
      acc[1][j] = WMMA_BF16(a1, bb, acc[1][j]);
    }
  }

  const int half = lane >> 4, ln = lane & 15;
  float qv[4];
#pragma unroll
  for (int j = 0; j < 4; ++j) qv[j] = qw[b * kM + c0 + 16 * j + ln];
#pragma unroll
  for (int i = 0; i < 2; ++i) {
    float cv[8];
#pragma unroll
    for (int e = 0; e < 8; ++e) cv[e] = cw[b * kN + r0 + 16 * i + half * 8 + e];
#pragma unroll
    for (int j = 0; j < 4; ++j) {
#pragma unroll
      for (int e = 0; e < 8; ++e) {
        const int row = r0 + 16 * i + half * 8 + e;
        const int col = c0 + 16 * j + ln;
        S[((size_t)b * kN + row) * kM + col] = acc[i][j][e] + cv[e] + qv[j];
      }
    }
  }
}

// ---------------------------------------------------------------------------
// K2: row softmax over m (q_mask) -> A bf16. One wave per (b,n) row, M=256.
// ---------------------------------------------------------------------------
__global__ void row_softmax_kernel(const float* __restrict__ S,
                                   const int* __restrict__ q_mask,
                                   bf16_t* __restrict__ Ab) {
  const int lane = threadIdx.x & 31;
  const int row = blockIdx.x * 8 + (threadIdx.x >> 5);  // [0, B*N)
  const int b = row >> 10;                              // row / kN
  const float* sr = S + (size_t)row * kM;
  const int* qm = q_mask + b * kM;

  const v4f* p = (const v4f*)(sr + lane * 8);
  const v4f lo = p[0], hi = p[1];
  float v[8];
#pragma unroll
  for (int i = 0; i < 4; ++i) { v[i] = lo[i]; v[4 + i] = hi[i]; }
#pragma unroll
  for (int i = 0; i < 8; ++i)
    if (!qm[lane * 8 + i]) v[i] = kNegBig;

  float mx = v[0];
#pragma unroll
  for (int i = 1; i < 8; ++i) mx = fmaxf(mx, v[i]);
  mx = wave_max(mx);

  float e[8], s = 0.f;
#pragma unroll
  for (int i = 0; i < 8; ++i) { e[i] = __expf(v[i] - mx); s += e[i]; }
  s = wave_sum(s);
  const float inv = 1.0f / s;

  v8bf o;
#pragma unroll
  for (int i = 0; i < 8; ++i) o[i] = (bf16_t)(e[i] * inv);
  *(v8bf*)(Ab + (size_t)row * kM + lane * 8) = o;
}

// ---------------------------------------------------------------------------
// K3: column softmax over n (c_mask) -> Bm bf16. One thread per (b,m) column,
// online max/sum; reads of S are coalesced across m. grid = B, block = M.
// ---------------------------------------------------------------------------
__global__ void col_softmax_kernel(const float* __restrict__ S,
                                   const int* __restrict__ c_mask,
                                   bf16_t* __restrict__ Bmb) {
  const int b = blockIdx.x;
  const int m = threadIdx.x;
  const float* sb = S + (size_t)b * kN * kM;
  const int* cm = c_mask + b * kN;

  float mx = -3.0e38f, sum = 0.f;
  for (int n = 0; n < kN; ++n) {
    if (cm[n]) {
      const float s = sb[(size_t)n * kM + m];
      const float nm = fmaxf(mx, s);
      sum = sum * __expf(mx - nm) + __expf(s - nm);
      mx = nm;
    }
  }
  const float inv = 1.0f / sum;
  bf16_t* ob = Bmb + (size_t)b * kN * kM;
  for (int n = 0; n < kN; ++n) {
    const float r = cm[n] ? __expf(sb[(size_t)n * kM + m] - mx) * inv : 0.f;
    ob[(size_t)n * kM + m] = (bf16_t)r;
  }
}

// ---------------------------------------------------------------------------
// K5: midT[b,d,m] = sum_j Bm[b,j,m] * context[b,j,d]
//   A = BmT [m, j] (K = N), B = CT [d, j]; output stored transposed in [d][m]
//   so the final GEMM's B-fragment loads are contiguous.
// Wave tile 16(m) x 64(d); waves = B*(M/16)*(D/64) = 4096.
// ---------------------------------------------------------------------------
__global__ __launch_bounds__(128) void mid_gemm_kernel(
    const bf16_t* __restrict__ BmT, const bf16_t* __restrict__ CT,
    bf16_t* __restrict__ midT) {
  const int lane = threadIdx.x & 31;
  const int wave = blockIdx.x * 4 + (threadIdx.x >> 5);
  const int b = wave >> 7;          // 128 waves per batch
  const int rem = wave & 127;
  const int mt = rem >> 3;          // 16 m-tiles
  const int dt = rem & 7;           // 8 d-tiles
  const bf16_t* A = BmT + (size_t)b * kM * kN;
  const bf16_t* Bc = CT + (size_t)b * kD * kN;

  v8f acc[4] = {};
  for (int kb = 0; kb < kN; kb += 32) {
    const v16bf a = load_a_frag(A, kN, mt * 16, kb, lane);
#pragma unroll
    for (int j = 0; j < 4; ++j) {
      const v16bf bb = load_b_frag(Bc, kN, dt * 64 + 16 * j, kb, lane);
      acc[j] = WMMA_BF16(a, bb, acc[j]);
    }
  }

  const int half = lane >> 4, ln = lane & 15;
  bf16_t* ob = midT + (size_t)b * kD * kM;
#pragma unroll
  for (int j = 0; j < 4; ++j) {
    const int d = dt * 64 + 16 * j + ln;
    v8bf o;
#pragma unroll
    for (int e = 0; e < 8; ++e) o[e] = (bf16_t)acc[j][e];
    *(v8bf*)(ob + (size_t)d * kM + mt * 16 + half * 8) = o;  // 16B store
  }
}

// ---------------------------------------------------------------------------
// K6: fused c2q = A @ Q and q2c = A @ mid (shared A fragments, K = M), plus
// output concat [context | c2q | context*c2q | context*q2c].
// Wave tile 16(n) x 64(d); waves = B*(N/16)*(D/64) = 16384.
// ---------------------------------------------------------------------------
__global__ __launch_bounds__(128) void final_gemm_kernel(
    const bf16_t* __restrict__ Ab, const bf16_t* __restrict__ QT,
    const bf16_t* __restrict__ midT, const float* __restrict__ context,
    float* __restrict__ out) {
  const int lane = threadIdx.x & 31;
  const int wave = blockIdx.x * 4 + (threadIdx.x >> 5);
  const int b = wave >> 9;          // 512 waves per batch
  const int rem = wave & 511;
  const int nt = rem >> 3;          // 64 n-tiles
  const int dt = rem & 7;           // 8 d-tiles
  const bf16_t* A = Ab + (size_t)b * kN * kM;
  const bf16_t* Bq = QT + (size_t)b * kD * kM;
  const bf16_t* Bm2 = midT + (size_t)b * kD * kM;

  v8f aq[4] = {}, am[4] = {};
  for (int kb = 0; kb < kM; kb += 32) {
    const v16bf a = load_a_frag(A, kM, nt * 16, kb, lane);
#pragma unroll
    for (int j = 0; j < 4; ++j) {
      const v16bf bq = load_b_frag(Bq, kM, dt * 64 + 16 * j, kb, lane);
      aq[j] = WMMA_BF16(a, bq, aq[j]);
      const v16bf bm = load_b_frag(Bm2, kM, dt * 64 + 16 * j, kb, lane);
      am[j] = WMMA_BF16(a, bm, am[j]);
    }
  }

  const int half = lane >> 4, ln = lane & 15;
#pragma unroll
  for (int j = 0; j < 4; ++j) {
#pragma unroll
    for (int e = 0; e < 8; ++e) {
      const int row = nt * 16 + half * 8 + e;
      const int col = dt * 64 + 16 * j + ln;
      const float ctx = context[((size_t)b * kN + row) * kD + col];
      float* o = out + ((size_t)b * kN + row) * (4 * kD);
      const float c2q = aq[j][e];
      const float q2c = am[j][e];
      o[col] = ctx;
      o[kD + col] = c2q;
      o[2 * kD + col] = ctx * c2q;
      o[3 * kD + col] = ctx * q2c;
    }
  }
}

// ---------------------------------------------------------------------------
// Host launcher
// ---------------------------------------------------------------------------
extern "C" void kernel_launch(void* const* d_in, const int* in_sizes, int n_in,
                              void* d_out, int out_size, void* d_ws, size_t ws_size,
                              hipStream_t stream) {
  (void)in_sizes; (void)n_in; (void)out_size; (void)ws_size;
  const float* context = (const float*)d_in[0];  // (B, N, D)
  const float* query   = (const float*)d_in[1];  // (B, M, D)
  const int*   c_mask  = (const int*)d_in[2];    // (B, N)
  const int*   q_mask  = (const int*)d_in[3];    // (B, M)
  const float* w       = (const float*)d_in[4];  // (3D, 1)
  float* out = (float*)d_out;                    // (B, N, 4D)

  char* ws = (char*)d_ws;
  size_t off = 0;
  auto take = [&](size_t bytes) -> char* {
    char* p = ws + off;
    off = (off + bytes + 255) & ~(size_t)255;
    return p;
  };
  float*  cw    = (float*)take((size_t)kB * kN * 4);        // C . w_c
  float*  qw    = (float*)take((size_t)kB * kM * 4);        // Q . w_q
  bf16_t* Cb    = (bf16_t*)take((size_t)kB * kN * kD * 2);  // bf16 C         [n][d]
  bf16_t* Qmb   = (bf16_t*)take((size_t)kB * kM * kD * 2);  // bf16 Q*diag wm [m][d]
  bf16_t* CTb   = (bf16_t*)take((size_t)kB * kD * kN * 2);  // bf16 C^T       [d][n]
  bf16_t* QTb   = (bf16_t*)take((size_t)kB * kD * kM * 2);  // bf16 Q^T       [d][m]
  float*  S     = (float*)take((size_t)kB * kN * kM * 4);   // scores (fits in L2)
  bf16_t* Ab    = (bf16_t*)take((size_t)kB * kN * kM * 2);  // softmax_m(S)   [n][m]
  bf16_t* Bmb   = (bf16_t*)take((size_t)kB * kN * kM * 2);  // softmax_n(S)   [n][m]
  bf16_t* BmTb  = (bf16_t*)take((size_t)kB * kM * kN * 2);  // Bm^T           [m][n]
  bf16_t* midTb = (bf16_t*)take((size_t)kB * kD * kM * 2);  // (Bm^T C)^T     [d][m]

  // Stage 0: bias dots + bf16 operand staging
  prep_ctx_kernel<<<kB * kN / 8, 256, 0, stream>>>(context, w, cw, Cb);
  prep_query_kernel<<<kB * kM / 8, 256, 0, stream>>>(query, w, qw, Qmb);
  transpose_to_bf16<float><<<dim3(kD / 32, kN / 32, kB), dim3(32, 8), 0, stream>>>(
      context, CTb, kN, kD);
  transpose_to_bf16<float><<<dim3(kD / 32, kM / 32, kB), dim3(32, 8), 0, stream>>>(
      query, QTb, kM, kD);

  // Stage 1: scores
  s_gemm_kernel<<<1024, 128, 0, stream>>>(Cb, Qmb, cw, qw, S);

  // Stage 2: dual softmax
  row_softmax_kernel<<<kB * kN / 8, 256, 0, stream>>>(S, q_mask, Ab);
  col_softmax_kernel<<<kB, kM, 0, stream>>>(S, c_mask, Bmb);
  transpose_to_bf16<bf16_t><<<dim3(kM / 32, kN / 32, kB), dim3(32, 8), 0, stream>>>(
      Bmb, BmTb, kN, kM);

  // Stage 3: mid = Bm^T @ C (stored transposed)
  mid_gemm_kernel<<<1024, 128, 0, stream>>>(BmTb, CTb, midTb);

  // Stage 4: fused c2q/q2c GEMMs + concat epilogue
  final_gemm_kernel<<<4096, 128, 0, stream>>>(Ab, QTb, midTb, context, out);
}
